// SwitchTransformerTabular_32186484917042
// MI455X (gfx1250) — matrix operand order repaired and checked by reading, not compiled, not run on hardware
//
#include <hip/hip_runtime.h>
#include <hip/hip_bf16.h>
#include <math.h>

// ---------------- problem constants ----------------
#define N_TOK 16384   // B*S
#define DIM   1024    // D
#define HID   4096    // H
#define NEXP  8       // E
#define CAP   4096    // capacity = 2.0 * N / E
#define LN_EPSF 1e-5f

// ---------------- GEMM tiling ----------------
#define BM 128
#define BN 128
#define BK 32
#define THREADS 256           // 8 waves of 32 (wave32)
// A-tile pitch: 40 bf16 = 80 B rows -> every 16B segment aligned for b128 async
// copies, and 20-DWORD row stride => 20*r mod 64 distinct for r=0..15 (no bank
// conflicts on fragment reads).
#define APITCH_A 40
// B-tile pitch: 34 bf16 = 17 DWORDs, gcd(17,64)=1 -> conflict-free.
#define APITCH_B 34

typedef __attribute__((ext_vector_type(16))) __bf16 v16bf;
typedef __attribute__((ext_vector_type(8)))  float  v8f;

// ---------------- bf16 helpers (bit-level, RNE) ----------------
__device__ __forceinline__ unsigned short f32_to_bf16_bits(float f) {
  union { float f; unsigned u; } c; c.f = f;
  unsigned u = c.u;
  return (unsigned short)((u + 0x7FFFu + ((u >> 16) & 1u)) >> 16);
}
__device__ __forceinline__ float bf16_bits_to_f32(unsigned short h) {
  union { unsigned u; float f; } c; c.u = ((unsigned)h) << 16;
  return c.f;
}

union Frag16 { v16bf v; unsigned u[8]; };
union Acc8   { v8f v; float f[8]; };

// A fragment: 16x32 bf16, per ISA 7.12.2: lane = M | (khalf<<4); kb = khalf*8,
// elems 0..7 = K[kb..kb+7], elems 8..15 = K[kb+16..kb+23]
template <int PITCH>
__device__ __forceinline__ v16bf load_frag_a(const unsigned short* lds, int rowBase) {
  int lane = threadIdx.x & 31;
  const unsigned short* p = lds + (rowBase + (lane & 15)) * PITCH;
  int kb = (lane >> 4) * 8;
  Frag16 f;
#pragma unroll
  for (int t = 0; t < 4; ++t) {
    f.u[t]     = *(const unsigned*)(p + kb + 2 * t);
    f.u[4 + t] = *(const unsigned*)(p + kb + 16 + 2 * t);
  }
  return f.v;
}
// B fragment: 32x16 bf16 staged K-transposed in LDS ([n][k]); kb = (lane>>4)*8,
// elems 0..7 = K[kb..kb+7], elems 8..15 = K[kb+16..kb+23], column = lane&15
__device__ __forceinline__ v16bf load_frag_b(const unsigned short* lds, int colBase) {
  int lane = threadIdx.x & 31;
  const unsigned short* p = lds + (colBase + (lane & 15)) * APITCH_B + (lane >> 4) * 8;
  Frag16 f;
#pragma unroll
  for (int t = 0; t < 4; ++t) {
    f.u[t]     = *(const unsigned*)(p + 2 * t);
    f.u[4 + t] = *(const unsigned*)(p + 16 + 2 * t);
  }
  return f.v;
}

__device__ __forceinline__ v8f wmma_bf16(v16bf a, v16bf b, v8f c) {
  return __builtin_amdgcn_wmma_f32_16x16x32_bf16(false, a, false, b, (short)0, c, false, false);
}

// ================= kernel 0: zero the token-output accumulator =================
__global__ void k_zero(float* __restrict__ p, long long n) {
  long long i = (long long)blockIdx.x * blockDim.x + threadIdx.x;
  long long st = (long long)gridDim.x * blockDim.x;
  for (; i < n; i += st) p[i] = 0.0f;
}

// ================= kernel 1: x_flat = x @ W_proj + b_proj  (WMMA bf16) =================
__global__ __launch_bounds__(THREADS)
void k_proj_gemm(const float* __restrict__ X, const float* __restrict__ W,
                 const float* __restrict__ bias, unsigned short* __restrict__ Y) {
  __shared__ alignas(16) unsigned short lA[BM * APITCH_A];
  __shared__ alignas(16) unsigned short lB[BN * APITCH_B];
  const int mBase = blockIdx.x * BM, nBase = blockIdx.y * BN;
  const int t = threadIdx.x, wave = t >> 5, lane = t & 31;

  v8f acc[8];
#pragma unroll
  for (int i = 0; i < 8; ++i) acc[i] = (v8f){0.f,0.f,0.f,0.f,0.f,0.f,0.f,0.f};

  for (int k0 = 0; k0 < DIM; k0 += BK) {
    if (k0 + BK < DIM)   // gfx1250 global_prefetch_b8 of next A tile
      __builtin_prefetch(X + (size_t)(mBase + (t >> 1)) * DIM + k0 + BK, 0, 1);
#pragma unroll
    for (int i = 0; i < (BM * BK) / THREADS; ++i) {
      int lin = i * THREADS + t, r = lin >> 5, c = lin & 31;
      lA[r * APITCH_A + c] = f32_to_bf16_bits(X[(size_t)(mBase + r) * DIM + k0 + c]);
    }
#pragma unroll
    for (int i = 0; i < (BN * BK) / THREADS; ++i) {
      int lin = i * THREADS + t, n = lin & (BN - 1), k = lin >> 7;
      lB[n * APITCH_B + k] = f32_to_bf16_bits(W[(size_t)(k0 + k) * DIM + nBase + n]);
    }
    __syncthreads();
    v16bf a = load_frag_a<APITCH_A>(lA, wave * 16);
#pragma unroll
    for (int nt = 0; nt < 8; ++nt)
      acc[nt] = wmma_bf16(a, load_frag_b(lB, nt * 16), acc[nt]);
    __syncthreads();
  }
  const int rBase = mBase + wave * 16 + ((lane >> 4) ? 8 : 0);
  const int col15 = lane & 15;
#pragma unroll
  for (int nt = 0; nt < 8; ++nt) {
    int col = nBase + nt * 16 + col15;
    float bv = bias[col];
    Acc8 a; a.v = acc[nt];
#pragma unroll
    for (int v = 0; v < 8; ++v)
      Y[(size_t)(rBase + v) * DIM + col] = f32_to_bf16_bits(a.f[v] + bv);
  }
}

// ================= kernel 2: router argmax (top-1; weight is analytically 1.0) ==========
__global__ __launch_bounds__(256)
void k_router(const unsigned short* __restrict__ Xf, const float* __restrict__ Wg,
              int* __restrict__ tokExp) {
  int token = blockIdx.x * 8 + (threadIdx.x >> 5);
  int lane = threadIdx.x & 31;
  float acc[NEXP];
#pragma unroll
  for (int e = 0; e < NEXP; ++e) acc[e] = 0.f;
  const unsigned short* xr = Xf + (size_t)token * DIM;
  for (int d = lane; d < DIM; d += 32) {
    float xv = bf16_bits_to_f32(xr[d]);
    const float* wr = Wg + (size_t)d * NEXP;
#pragma unroll
    for (int e = 0; e < NEXP; ++e) acc[e] += xv * wr[e];
  }
#pragma unroll
  for (int e = 0; e < NEXP; ++e)
#pragma unroll
    for (int off = 16; off > 0; off >>= 1) acc[e] += __shfl_down(acc[e], off, 32);
  if (lane == 0) {
    int best = 0; float bv = acc[0];
#pragma unroll
    for (int e = 1; e < NEXP; ++e) if (acc[e] > bv) { bv = acc[e]; best = e; }
    tokExp[token] = best;   // softmax is monotone; ties -> lowest index, same as top_k
  }
}

// ===== kernel 3: deterministic per-expert capacity compaction (ballot prefix-sum) ======
__global__ __launch_bounds__(256)
void k_select(const int* __restrict__ tokExp, int* __restrict__ expIdx, int* __restrict__ expCnt) {
  int e = blockIdx.x;
  __shared__ int warpCnt[8];
  int t = threadIdx.x, wave = t >> 5, lane = t & 31;
  unsigned ltmask = (lane == 0) ? 0u : ((1u << lane) - 1u);
  int base = 0;
  for (int c0 = 0; c0 < N_TOK; c0 += 256) {
    int i = c0 + t;
    bool pred = (tokExp[i] == e);
    unsigned mask = (unsigned)__ballot(pred);   // wave32: low 32 bits
    if (lane == 0) warpCnt[wave] = __popc(mask);
    __syncthreads();
    int excl = 0, total = 0;
#pragma unroll
    for (int w = 0; w < 8; ++w) { int c = warpCnt[w]; total += c; if (w < wave) excl += c; }
    int slot = base + excl + __popc(mask & ltmask);
    if (pred && slot < CAP) expIdx[e * CAP + slot] = i;
    base += total;
    __syncthreads();
  }
  if (t == 0) expCnt[e] = base < CAP ? base : CAP;
}

// ============ kernel 4: expand expert slots to row->token map (replaces gather) ========
__global__ __launch_bounds__(256)
void k_rowtok(const int* __restrict__ expIdx, const int* __restrict__ expCnt,
              int* __restrict__ rowTok) {
  int row = blockIdx.x * 256 + threadIdx.x;   // e * CAP + c
  int e = row >> 12, c = row & (CAP - 1);
  rowTok[row] = (c < expCnt[e]) ? expIdx[row] : -1;
}

// ====== kernel 5: FFN1 h = x_flat[rowTok] @ W1[e] + b1 (gather fused via async A) ======
// A-tile is already bf16 in global memory -> stream it straight to LDS with the
// gfx1250 async-copy path (ASYNCcnt), no VGPR round-trip, gather fused into the
// per-lane async address.
__global__ __launch_bounds__(THREADS)
void k_ffn1_gemm(const unsigned short* __restrict__ Xf, const float* __restrict__ W1,
                 const float* __restrict__ b1, const int* __restrict__ rowTok,
                 unsigned short* __restrict__ Hout) {
  __shared__ alignas(16) unsigned short lA[BM * APITCH_A];
  __shared__ alignas(16) unsigned short lB[BN * APITCH_B];
  const int e = blockIdx.z;
  const float* W = W1 + (size_t)e * DIM * HID;
  const float* bias = b1 + (size_t)e * HID;
  unsigned short* Y = Hout + (size_t)e * CAP * HID;
  const int rowOff = e * CAP;
  const int mBase = blockIdx.x * BM, nBase = blockIdx.y * BN;
  const int t = threadIdx.x, wave = t >> 5, lane = t & 31;

  // Per-thread fixed (row, 16B-segment) assignment for the async A staging:
  // 128 rows * 4 segs = 512 b128 chunks = 2 per thread.
  const int rSeg = t & 3;             // which 16B segment of the 64B row
  const int r0 = t >> 2;              // rows r0 and r0+64
  const unsigned ldsBaseA = (unsigned)(size_t)(&lA[0]);
  int tokR[2];
  unsigned ldsAddr[2];
  unsigned gBase[2];
#pragma unroll
  for (int i = 0; i < 2; ++i) {
    int r = r0 + i * 64;
    tokR[i] = rowTok[rowOff + mBase + r];
    ldsAddr[i] = ldsBaseA + (unsigned)(r * (APITCH_A * 2) + rSeg * 16);
    gBase[i] = (unsigned)tokR[i] * (unsigned)(DIM * 2) + (unsigned)(rSeg * 16);
  }

  v8f acc[8];
#pragma unroll
  for (int i = 0; i < 8; ++i) acc[i] = (v8f){0.f,0.f,0.f,0.f,0.f,0.f,0.f,0.f};

  for (int k0 = 0; k0 < DIM; k0 += BK) {
    // ---- A tile: async global->LDS copy (bf16, indirect through rowTok) ----
#pragma unroll
    for (int i = 0; i < 2; ++i) {
      if (tokR[i] >= 0) {
        unsigned goff = gBase[i] + (unsigned)(k0 * 2);
        asm volatile("global_load_async_to_lds_b128 %0, %1, %2"
                     :: "v"(ldsAddr[i]), "v"(goff), "s"(Xf)
                     : "memory");
      } else {
        uint4 z = {0u, 0u, 0u, 0u};
        *reinterpret_cast<uint4*>(reinterpret_cast<char*>(lA) +
                                  (size_t)(r0 + i * 64) * (APITCH_A * 2) + rSeg * 16) = z;
      }
    }
    // ---- B tile: f32 -> bf16 conversion staging (overlaps with async A) ----
    if (k0 + BK < DIM)
      __builtin_prefetch(W + (size_t)(k0 + BK + (t >> 3)) * HID + nBase, 0, 1);
#pragma unroll
    for (int i = 0; i < (BN * BK) / THREADS; ++i) {
      int lin = i * THREADS + t, n = lin & (BN - 1), k = lin >> 7;
      lB[n * APITCH_B + k] = f32_to_bf16_bits(W[(size_t)(k0 + k) * HID + nBase + n]);
    }
    asm volatile("s_wait_asynccnt 0x0" ::: "memory");  // my async LDS writes landed
    __syncthreads();                                   // everyone's writes visible
    v16bf a = load_frag_a<APITCH_A>(lA, wave * 16);
#pragma unroll
    for (int nt = 0; nt < 8; ++nt)
      acc[nt] = wmma_bf16(a, load_frag_b(lB, nt * 16), acc[nt]);
    __syncthreads();
  }
  const int rBase = mBase + wave * 16 + ((lane >> 4) ? 8 : 0);
  const int col15 = lane & 15;
#pragma unroll
  for (int nt = 0; nt < 8; ++nt) {
    int col = nBase + nt * 16 + col15;
    float bv = bias[col];
    Acc8 a; a.v = acc[nt];
#pragma unroll
    for (int v = 0; v < 8; ++v)
      Y[(size_t)(rBase + v) * HID + col] = f32_to_bf16_bits(a.f[v] + bv);
  }
}

// ================= kernel 6: per-row LN stats over H ===================================
__global__ __launch_bounds__(256)
void k_lnstats(const unsigned short* __restrict__ Hh, float* __restrict__ mu,
               float* __restrict__ rs) {
  __shared__ float red[256];
  int row = blockIdx.x;
  const unsigned short* hr = Hh + (size_t)row * HID;
  float s = 0.f;
  for (int i = threadIdx.x; i < HID; i += 256) s += bf16_bits_to_f32(hr[i]);
  red[threadIdx.x] = s; __syncthreads();
  for (int off = 128; off > 0; off >>= 1) {
    if (threadIdx.x < off) red[threadIdx.x] += red[threadIdx.x + off];
    __syncthreads();
  }
  float m = red[0] * (1.0f / HID);
  __syncthreads();
  float v = 0.f;
  for (int i = threadIdx.x; i < HID; i += 256) {
    float d = bf16_bits_to_f32(hr[i]) - m; v += d * d;
  }
  red[threadIdx.x] = v; __syncthreads();
  for (int off = 128; off > 0; off >>= 1) {
    if (threadIdx.x < off) red[threadIdx.x] += red[threadIdx.x + off];
    __syncthreads();
  }
  if (threadIdx.x == 0) { mu[row] = m; rs[row] = rsqrtf(red[0] * (1.0f / HID) + LN_EPSF); }
}

// ====== kernel 7: FFN2 out = gelu(LN(h)) @ W2[e] + b2, fused scatter (weight==1) =======
__global__ __launch_bounds__(THREADS)
void k_ffn2_gemm(const unsigned short* __restrict__ Hh, const float* __restrict__ W2,
                 const float* __restrict__ b2, const float* __restrict__ ln1g,
                 const float* __restrict__ ln1b, const float* __restrict__ mu,
                 const float* __restrict__ rs, const int* __restrict__ rowTok,
                 float* __restrict__ Out) {
  __shared__ alignas(16) unsigned short lA[BM * APITCH_A];
  __shared__ alignas(16) unsigned short lB[BN * APITCH_B];
  const int e = blockIdx.z;
  const unsigned short* A = Hh + (size_t)e * CAP * HID;
  const float* W = W2 + (size_t)e * HID * DIM;
  const float* bias = b2 + (size_t)e * DIM;
  const float* g_ = ln1g + (size_t)e * HID;
  const float* b_ = ln1b + (size_t)e * HID;
  const int rowOff = e * CAP;
  const int mBase = blockIdx.x * BM, nBase = blockIdx.y * BN;
  const int t = threadIdx.x, wave = t >> 5, lane = t & 31;

  v8f acc[8];
#pragma unroll
  for (int i = 0; i < 8; ++i) acc[i] = (v8f){0.f,0.f,0.f,0.f,0.f,0.f,0.f,0.f};

  for (int k0 = 0; k0 < HID; k0 += BK) {
    if (k0 + BK < HID)
      __builtin_prefetch(A + (size_t)(mBase + (t >> 1)) * HID + k0 + BK, 0, 1);
#pragma unroll
    for (int i = 0; i < (BM * BK) / THREADS; ++i) {
      int lin = i * THREADS + t, r = lin >> 5, c = lin & 31;
      int rg = mBase + r;
      float x  = bf16_bits_to_f32(A[(size_t)rg * HID + k0 + c]);
      float xn = (x - mu[rowOff + rg]) * rs[rowOff + rg] * g_[k0 + c] + b_[k0 + c];
      float act = 0.5f * xn * (1.0f + erff(xn * 0.70710678f));  // exact gelu
      lA[r * APITCH_A + c] = f32_to_bf16_bits(act);
    }
#pragma unroll
    for (int i = 0; i < (BN * BK) / THREADS; ++i) {
      int lin = i * THREADS + t, n = lin & (BN - 1), k = lin >> 7;
      lB[n * APITCH_B + k] = f32_to_bf16_bits(W[(size_t)(k0 + k) * DIM + nBase + n]);
    }
    __syncthreads();
    v16bf a = load_frag_a<APITCH_A>(lA, wave * 16);
#pragma unroll
    for (int nt = 0; nt < 8; ++nt)
      acc[nt] = wmma_bf16(a, load_frag_b(lB, nt * 16), acc[nt]);
    __syncthreads();
  }
  const int rBase = mBase + wave * 16 + ((lane >> 4) ? 8 : 0);
  const int col15 = lane & 15;
  Acc8 au[8];
#pragma unroll
  for (int nt = 0; nt < 8; ++nt) au[nt].v = acc[nt];
#pragma unroll
  for (int v = 0; v < 8; ++v) {
    int token = rowTok[rowOff + rBase + v];
    if (token < 0) continue;                 // dropped / empty slot
    float* orow = Out + (size_t)token * DIM;
#pragma unroll
    for (int nt = 0; nt < 8; ++nt) {
      int col = nBase + nt * 16 + col15;
      orow[col] = au[nt].f[v] + bias[col];   // top-1 routing weight == 1.0 exactly
    }
  }
}

// ================= kernel 8: final LayerNorm over D ====================================
__global__ __launch_bounds__(256)
void k_final_ln(const float* __restrict__ O, const float* __restrict__ g,
                const float* __restrict__ b, float* __restrict__ out) {
  __shared__ float red[256];
  int row = blockIdx.x;
  const float* xr = O + (size_t)row * DIM;
  float s = 0.f;
  for (int i = threadIdx.x; i < DIM; i += 256) s += xr[i];
  red[threadIdx.x] = s; __syncthreads();
  for (int off = 128; off > 0; off >>= 1) {
    if (threadIdx.x < off) red[threadIdx.x] += red[threadIdx.x + off];
    __syncthreads();
  }
  float m = red[0] * (1.0f / DIM);
  __syncthreads();
  float v = 0.f;
  for (int i = threadIdx.x; i < DIM; i += 256) { float d = xr[i] - m; v += d * d; }
  red[threadIdx.x] = v; __syncthreads();
  for (int off = 128; off > 0; off >>= 1) {
    if (threadIdx.x < off) red[threadIdx.x] += red[threadIdx.x + off];
    __syncthreads();
  }
  float rsg = rsqrtf(red[0] * (1.0f / DIM) + LN_EPSF);
  for (int i = threadIdx.x; i < DIM; i += 256)
    out[(size_t)row * DIM + i] = (xr[i] - m) * rsg * g[i] + b[i];
}

// ======================================================================================
extern "C" void kernel_launch(void* const* d_in, const int* in_sizes, int n_in,
                              void* d_out, int out_size, void* d_ws, size_t ws_size,
                              hipStream_t stream) {
  const float* x      = (const float*)d_in[0];
  const float* W_proj = (const float*)d_in[1];
  const float* b_proj = (const float*)d_in[2];
  const float* W_gate = (const float*)d_in[3];
  const float* W1     = (const float*)d_in[4];
  const float* b1     = (const float*)d_in[5];
  const float* ln1_g  = (const float*)d_in[6];
  const float* ln1_b  = (const float*)d_in[7];
  const float* W2     = (const float*)d_in[8];
  const float* b2     = (const float*)d_in[9];
  const float* ln_g   = (const float*)d_in[10];
  const float* ln_b   = (const float*)d_in[11];
  float* out = (float*)d_out;

  // workspace carve-up
  char* ws = (char*)d_ws;
  size_t off = 0;
  auto carve = [&](size_t bytes) -> void* {
    void* p = ws + off;
    off += (bytes + 255) & ~(size_t)255;
    return p;
  };
  unsigned short* xflat  = (unsigned short*)carve((size_t)N_TOK * DIM * 2);       // 32 MB
  unsigned short* hbuf   = (unsigned short*)carve((size_t)NEXP * CAP * HID * 2);  // 256 MB
  float* outputs         = (float*)carve((size_t)N_TOK * DIM * 4);                // 64 MB
  float* mu              = (float*)carve((size_t)NEXP * CAP * 4);
  float* rs              = (float*)carve((size_t)NEXP * CAP * 4);
  int*   tokExp          = (int*)carve((size_t)N_TOK * 4);
  int*   expIdx          = (int*)carve((size_t)NEXP * CAP * 4);
  int*   expCnt          = (int*)carve((size_t)NEXP * 4);
  int*   rowTok          = (int*)carve((size_t)NEXP * CAP * 4);
  (void)ws_size; (void)in_sizes; (void)n_in; (void)out_size;

  k_zero<<<4096, 256, 0, stream>>>(outputs, (long long)N_TOK * DIM);
  k_proj_gemm<<<dim3(N_TOK / BM, DIM / BN), THREADS, 0, stream>>>(x, W_proj, b_proj, xflat);
  k_router<<<N_TOK / 8, 256, 0, stream>>>(xflat, W_gate, tokExp);
  k_select<<<NEXP, 256, 0, stream>>>(tokExp, expIdx, expCnt);
  k_rowtok<<<NEXP * CAP / 256, 256, 0, stream>>>(expIdx, expCnt, rowTok);
  k_ffn1_gemm<<<dim3(CAP / BM, HID / BN, NEXP), THREADS, 0, stream>>>(
      xflat, W1, b1, rowTok, hbuf);
  k_lnstats<<<NEXP * CAP, 256, 0, stream>>>(hbuf, mu, rs);
  k_ffn2_gemm<<<dim3(CAP / BM, DIM / BN, NEXP), THREADS, 0, stream>>>(
      hbuf, W2, b2, ln1_g, ln1_b, mu, rs, rowTok, outputs);
  k_final_ln<<<N_TOK, 256, 0, stream>>>(outputs, ln_g, ln_b, out);
}
